// VAE_AttentionBlock_38474317037665
// MI455X (gfx1250) — matrix operand order repaired and checked by reading, not compiled, not run on hardware
//
#include <hip/hip_runtime.h>
#include <hip/hip_bf16.h>
#include <math.h>

// ---------------- problem constants ----------------
#define N_BATCH 4
#define C_DIM   512
#define HW      4096
#define GROUPS  32
#define CPG     16          // channels per group
#define D3      1536        // 3*C
#define QK_SCALE 0.044194173824159216f   // 1/sqrt(512)

typedef __attribute__((ext_vector_type(16))) _Float16 v16h;
typedef __attribute__((ext_vector_type(8)))  _Float16 v8h;
typedef __attribute__((ext_vector_type(8)))  float    v8f;

// ---------------- WMMA helpers (wave32, 16x16x32 f16 -> f32) ----------------
__device__ __forceinline__ v8f wmma16(v16h a, v16h b, v8f c) {
    // D = A(16x32) * B(32x16) + C(16x16)
    return __builtin_amdgcn_wmma_f32_16x16x32_f16(
        /*neg_a=*/false, a, /*neg_b=*/false, b,
        /*c_mod=*/(short)0, c, /*reuse_a=*/false, /*reuse_b=*/false);
}

// A fragment: 16x32 (MxK). Lane L holds row M = row0 + L%16, half h = L/16.
// VGPR0..3 hold K = h*8 .. h*8+7 ; VGPR4..7 hold K = 16+h*8 .. 16+h*8+7
__device__ __forceinline__ v16h load_a_frag(const _Float16* base, int ld, int row0, int k0) {
    int lane = threadIdx.x & 31;
    const _Float16* p = base + (size_t)(row0 + (lane & 15)) * ld + k0 + ((lane >> 4) << 3);
    v8h lo = *(const v8h*)(p);        // K = h*8 .. +7
    v8h hi = *(const v8h*)(p + 16);   // K = 16+h*8 .. +7
    v16h f;
#pragma unroll
    for (int i = 0; i < 8; ++i) { f[i] = lo[i]; f[i + 8] = hi[i]; }
    return f;
}

// B fragment: 32x16 (KxN), supplied as rows of B^T (row = output column).
// Lane L holds column N = col0 + L%16; elements 0..7 hold K = h*8 .. +7, elements 8..15 hold K = 16+h*8 .. +7 (h = L/16).
__device__ __forceinline__ v16h load_b_frag(const _Float16* base, int ld, int col0, int k0) {
    int lane = threadIdx.x & 31;
    const _Float16* p = base + (size_t)(col0 + (lane & 15)) * ld + k0 + ((lane >> 4) << 3);
    v8h lo = *(const v8h*)(p);
    v8h hi = *(const v8h*)(p + 16);
    v16h f;
#pragma unroll
    for (int i = 0; i < 8; ++i) { f[i] = lo[i]; f[i + 8] = hi[i]; }
    return f;
}

// ---------------- kernel 0: weights f32 -> f16 ----------------
__global__ void cvt_weights_kernel(const float* __restrict__ w_in,
                                   const float* __restrict__ w_out,
                                   _Float16* __restrict__ wi_h,
                                   _Float16* __restrict__ wo_h) {
    int i = blockIdx.x * 256 + threadIdx.x;
    if (i < D3 * C_DIM)    wi_h[i] = (_Float16)w_in[i];
    if (i < C_DIM * C_DIM) wo_h[i] = (_Float16)w_out[i];
}

// ---------------- kernel 1: GroupNorm + transpose to [n, s, c] f16 ----------------
__global__ void gn_kernel(const float* __restrict__ x,
                          const float* __restrict__ gs,
                          const float* __restrict__ gb,
                          _Float16* __restrict__ xn) {
    __shared__ float rs[256], rs2[256];
    int n = blockIdx.x >> 5;       // GROUPS == 32
    int g = blockIdx.x & 31;
    const float* xp = x + ((size_t)n * C_DIM + g * CPG) * HW;
    int tid = threadIdx.x;
    float s = 0.f, s2 = 0.f;
    for (int i = tid; i < CPG * HW; i += 256) {
        float v = xp[i]; s += v; s2 += v * v;
    }
    rs[tid] = s; rs2[tid] = s2;
    __syncthreads();
    for (int st = 128; st > 0; st >>= 1) {
        if (tid < st) { rs[tid] += rs[tid + st]; rs2[tid] += rs2[tid + st]; }
        __syncthreads();
    }
    const float inv = 1.f / (float)(CPG * HW);
    float mean = rs[0] * inv;
    float var  = rs2[0] * inv - mean * mean;
    float rstd = rsqrtf(var + 1e-5f);
    for (int i = tid; i < CPG * HW; i += 256) {
        int cl = i >> 12;            // i / HW
        int sp = i & (HW - 1);       // i % HW
        int c  = g * CPG + cl;
        float v = (xp[i] - mean) * rstd * gs[c] + gb[c];
        xn[((size_t)n * HW + sp) * C_DIM + c] = (_Float16)v;
    }
}

// ---------------- kernel 2: fused QKV GEMM ----------------
// [4096 x 512] (f16) x w_in^T [512 x 1536] -> Q (scaled, [s][c]), K ([s][c]), Vt ([c][s])
__global__ void qkv_kernel(const _Float16* __restrict__ xn,
                           const _Float16* __restrict__ wi_h,
                           const float* __restrict__ b_in,
                           _Float16* __restrict__ qh,
                           _Float16* __restrict__ kh,
                           _Float16* __restrict__ vt) {
    int n  = blockIdx.x / (HW / 64);
    int m0 = (blockIdx.x % (HW / 64)) * 64;
    int d0 = blockIdx.y * 64;
    int w = threadIdx.x >> 5, lane = threadIdx.x & 31;
    int mw = m0 + (w >> 1) * 32;
    int dw = d0 + (w & 1) * 32;
    const _Float16* A = xn + (size_t)n * HW * C_DIM;

    v8f acc[2][2] = {};
#pragma unroll 4
    for (int k0 = 0; k0 < C_DIM; k0 += 32) {
        v16h a0 = load_a_frag(A, C_DIM, mw,      k0);
        v16h a1 = load_a_frag(A, C_DIM, mw + 16, k0);
        v16h b0 = load_b_frag(wi_h, C_DIM, dw,      k0);
        v16h b1 = load_b_frag(wi_h, C_DIM, dw + 16, k0);
        acc[0][0] = wmma16(a0, b0, acc[0][0]);
        acc[0][1] = wmma16(a0, b1, acc[0][1]);
        acc[1][0] = wmma16(a1, b0, acc[1][0]);
        acc[1][1] = wmma16(a1, b1, acc[1][1]);
    }
    int h = lane >> 4, nn = lane & 15;
#pragma unroll
    for (int i = 0; i < 2; ++i)
#pragma unroll
    for (int j = 0; j < 2; ++j) {
        int d = dw + j * 16 + nn;
        float bias = b_in[d];
#pragma unroll
        for (int r = 0; r < 8; ++r) {
            int sidx = mw + i * 16 + r + 8 * h;
            float v = acc[i][j][r] + bias;
            if (d < C_DIM) {
                qh[((size_t)n * HW + sidx) * C_DIM + d] = (_Float16)(v * QK_SCALE);
            } else if (d < 2 * C_DIM) {
                kh[((size_t)n * HW + sidx) * C_DIM + (d - C_DIM)] = (_Float16)v;
            } else {
                vt[((size_t)n * C_DIM + (d - 2 * C_DIM)) * HW + sidx] = (_Float16)v;
            }
        }
    }
}

// ---------------- kernel 3: flash attention ----------------
// block: 256 threads (8 waves), handles 32 Q rows; streams K/V in 64-wide tiles.
#define TQ 32
#define TK 64
__global__ void attn_kernel(const _Float16* __restrict__ qh,
                            const _Float16* __restrict__ kh,
                            const _Float16* __restrict__ vt,
                            _Float16* __restrict__ ao) {
    __shared__ __align__(16) float    Sld[TQ * TK];
    __shared__ __align__(16) _Float16 Pld[TQ * TK];
    __shared__ float red[256];
    __shared__ float mrow[TQ], lrow[TQ], resc[TQ];

    int n  = blockIdx.x / (HW / TQ);
    int m0 = (blockIdx.x % (HW / TQ)) * TQ;
    int tid = threadIdx.x, w = tid >> 5, lane = tid & 31;
    int mi = w >> 2;   // 0..1 : which 16-row half of the Q strip
    int qi = w & 3;    // 0..3 : t-subtile (stage A) / 128-dim slice (stage C)
    int h = lane >> 4, nn = lane & 15;

    const _Float16* Q = qh + (size_t)n * HW * C_DIM;
    const _Float16* K = kh + (size_t)n * HW * C_DIM;
    const _Float16* V = vt + (size_t)n * C_DIM * HW;   // [c][t]

    if (tid < TQ) { mrow[tid] = -INFINITY; lrow[tid] = 0.f; }
    v8f oacc[8];
#pragma unroll
    for (int t = 0; t < 8; ++t) oacc[t] = (v8f){};
    __syncthreads();

    for (int t0 = 0; t0 < HW; t0 += TK) {
        // ---- stage A: S(16x16) = Q_strip x K_tile^T  (Q pre-scaled) ----
        v8f sacc = {};
#pragma unroll 4
        for (int k0 = 0; k0 < C_DIM; k0 += 32) {
            v16h a = load_a_frag(Q, C_DIM, m0 + mi * 16, k0);
            v16h b = load_b_frag(K, C_DIM, t0 + qi * 16, k0);
            sacc = wmma16(a, b, sacc);
        }
#pragma unroll
        for (int r = 0; r < 8; ++r)
            Sld[(mi * 16 + r + 8 * h) * TK + qi * 16 + nn] = sacc[r];
        __syncthreads();

        // ---- stage B: online softmax (f32), 8 threads per row ----
        int row = tid >> 3;
        int cb  = (tid & 7) * 8;
        float lm = -INFINITY;
#pragma unroll
        for (int j = 0; j < 8; ++j) lm = fmaxf(lm, Sld[row * TK + cb + j]);
        red[tid] = lm;
        __syncthreads();
        if ((tid & 7) == 0) {
            float m = red[tid];
#pragma unroll
            for (int j = 1; j < 8; ++j) m = fmaxf(m, red[tid + j]);
            float mold = mrow[row];
            float mnew = fmaxf(mold, m);
            mrow[row] = mnew;
            resc[row] = __expf(mold - mnew);   // 0 on first tile
        }
        __syncthreads();
        float mnew = mrow[row];
        float ls = 0.f;
#pragma unroll
        for (int j = 0; j < 8; ++j) {
            float p = __expf(Sld[row * TK + cb + j] - mnew);
            Pld[row * TK + cb + j] = (_Float16)p;
            ls += p;
        }
        red[tid] = ls;
        __syncthreads();
        if ((tid & 7) == 0) {
            float s = 0.f;
#pragma unroll
            for (int j = 0; j < 8; ++j) s += red[tid + j];
            lrow[row] = lrow[row] * resc[row] + s;
        }

        // ---- stage C: rescale accumulators, add P(16x64) x V^T(64x128-slice) ----
        float f[8];
#pragma unroll
        for (int r = 0; r < 8; ++r) f[r] = resc[mi * 16 + r + 8 * h];
#pragma unroll
        for (int t = 0; t < 8; ++t)
#pragma unroll
            for (int r = 0; r < 8; ++r) oacc[t][r] *= f[r];

        v16h a0 = load_a_frag(Pld, TK, mi * 16, 0);
        v16h a1 = load_a_frag(Pld, TK, mi * 16, 32);
#pragma unroll
        for (int t = 0; t < 8; ++t) {
            int c0 = qi * 128 + t * 16;
            v16h b0 = load_b_frag(V, HW, c0, t0);
            v16h b1 = load_b_frag(V, HW, c0, t0 + 32);
            oacc[t] = wmma16(a0, b0, oacc[t]);
            oacc[t] = wmma16(a1, b1, oacc[t]);
        }
        __syncthreads();
    }

    // ---- finalize: divide by l, store [n][s][c] f16 ----
#pragma unroll
    for (int t = 0; t < 8; ++t) {
#pragma unroll
        for (int r = 0; r < 8; ++r) {
            int srow = m0 + mi * 16 + r + 8 * h;
            float linv = 1.f / lrow[mi * 16 + r + 8 * h];
            ao[((size_t)n * HW + srow) * C_DIM + qi * 128 + t * 16 + nn] =
                (_Float16)(oacc[t][r] * linv);
        }
    }
}

// ---------------- kernel 4: out projection + bias + residual + NCHW store ----------------
__global__ void proj_kernel(const _Float16* __restrict__ ao,
                            const _Float16* __restrict__ wo_h,
                            const float* __restrict__ b_out,
                            const float* __restrict__ x,
                            float* __restrict__ out) {
    int n  = blockIdx.x / (HW / 64);
    int m0 = (blockIdx.x % (HW / 64)) * 64;
    int c0 = blockIdx.y * 64;
    int w = threadIdx.x >> 5, lane = threadIdx.x & 31;
    int mw = m0 + (w >> 1) * 32;
    int cw = c0 + (w & 1) * 32;
    const _Float16* A = ao + (size_t)n * HW * C_DIM;

    v8f acc[2][2] = {};
#pragma unroll 4
    for (int k0 = 0; k0 < C_DIM; k0 += 32) {
        v16h a0 = load_a_frag(A, C_DIM, mw,      k0);
        v16h a1 = load_a_frag(A, C_DIM, mw + 16, k0);
        v16h b0 = load_b_frag(wo_h, C_DIM, cw,      k0);
        v16h b1 = load_b_frag(wo_h, C_DIM, cw + 16, k0);
        acc[0][0] = wmma16(a0, b0, acc[0][0]);
        acc[0][1] = wmma16(a0, b1, acc[0][1]);
        acc[1][0] = wmma16(a1, b0, acc[1][0]);
        acc[1][1] = wmma16(a1, b1, acc[1][1]);
    }
    int h = lane >> 4, nn = lane & 15;
#pragma unroll
    for (int i = 0; i < 2; ++i)
#pragma unroll
    for (int j = 0; j < 2; ++j) {
        int c = cw + j * 16 + nn;
        float bias = b_out[c];
#pragma unroll
        for (int r = 0; r < 8; ++r) {
            int sidx = mw + i * 16 + r + 8 * h;
            size_t oi = ((size_t)n * C_DIM + c) * HW + sidx;
            out[oi] = acc[i][j][r] + bias + x[oi];
        }
    }
}

// ---------------- launcher ----------------
extern "C" void kernel_launch(void* const* d_in, const int* in_sizes, int n_in,
                              void* d_out, int out_size, void* d_ws, size_t ws_size,
                              hipStream_t stream) {
    const float* x        = (const float*)d_in[0];
    const float* gn_scale = (const float*)d_in[1];
    const float* gn_bias  = (const float*)d_in[2];
    const float* w_in     = (const float*)d_in[3];
    const float* b_in     = (const float*)d_in[4];
    const float* w_out    = (const float*)d_in[5];
    const float* b_out    = (const float*)d_in[6];
    float* out = (float*)d_out;

    char* ws = (char*)d_ws;
    size_t off = 0;
    auto carve = [&](size_t bytes) -> char* {
        char* p = ws + off;
        off = (off + bytes + 255) & ~(size_t)255;
        return p;
    };
    _Float16* wi_h = (_Float16*)carve((size_t)D3 * C_DIM * 2);
    _Float16* wo_h = (_Float16*)carve((size_t)C_DIM * C_DIM * 2);
    _Float16* xn   = (_Float16*)carve((size_t)N_BATCH * HW * C_DIM * 2);
    _Float16* qh   = (_Float16*)carve((size_t)N_BATCH * HW * C_DIM * 2);
    _Float16* kh   = (_Float16*)carve((size_t)N_BATCH * HW * C_DIM * 2);
    _Float16* vt   = (_Float16*)carve((size_t)N_BATCH * C_DIM * HW * 2);
    _Float16* ao   = (_Float16*)carve((size_t)N_BATCH * HW * C_DIM * 2);
    (void)ws_size; (void)in_sizes; (void)n_in; (void)out_size;

    cvt_weights_kernel<<<(D3 * C_DIM + 255) / 256, 256, 0, stream>>>(w_in, w_out, wi_h, wo_h);
    gn_kernel<<<N_BATCH * GROUPS, 256, 0, stream>>>(x, gn_scale, gn_bias, xn);
    qkv_kernel<<<dim3(N_BATCH * (HW / 64), D3 / 64), 128, 0, stream>>>(xn, wi_h, b_in, qh, kh, vt);
    attn_kernel<<<N_BATCH * (HW / TQ), 256, 0, stream>>>(qh, kh, vt, ao);
    proj_kernel<<<dim3(N_BATCH * (HW / 64), C_DIM / 64), 128, 0, stream>>>(ao, wo_h, b_out, x, out);
}
